// Model_2619930051402
// MI455X (gfx1250) — matrix operand order repaired and checked by reading, not compiled, not run on hardware
//
#include <hip/hip_runtime.h>
#include <hip/hip_bf16.h>

#define NT 2048
#define NH 8
#define QM 4
#define DH 64
#define SW 128
#define SM_SCALE 0.1767766952966369f
#define LOG2E    1.4426950408889634f

typedef __attribute__((ext_vector_type(16))) __bf16    v16bf;
typedef __attribute__((ext_vector_type(2)))  __bf16    v2bf;
typedef __attribute__((ext_vector_type(8)))  unsigned  v8u;
typedef __attribute__((ext_vector_type(8)))  float     v8f;

// native fp32 -> bf16 (RNE), single element
__device__ __forceinline__ unsigned short f2bf(float f) {
  return __builtin_bit_cast(unsigned short, (__bf16)f);
}

// native fp32 pair -> packed 2x bf16 in one dword (encourages v_cvt_pk_bf16_f32)
__device__ __forceinline__ unsigned pk2bf(float x, float y) {
  v2bf p = { (__bf16)x, (__bf16)y };
  return __builtin_bit_cast(unsigned, p);
}

__device__ __forceinline__ unsigned ld2(const unsigned short* p) {
  return *(const unsigned*)p;  // two packed bf16, 4B-aligned by construction
}

// A-fragment K index for 16-bit WMMA A (16x32): lanes 0-15 hold K {0..7,16..23},
// lanes 16-31 hold K {8..15,24..31}, pairs per VGPR (ISA 7.12.2).
__device__ __forceinline__ int a_kidx(int i, int half) {
  return ((i & 4) << 2) + 8 * half + 2 * (i & 3);
}

// Max all-reduce across the 16 lanes of each wave half, pure VALU (DPP16):
// quad_perm xor1, quad_perm xor2, row_half_mirror, row_mirror.
__device__ __forceinline__ float dpp_max16(float v) {
  int x = __float_as_int(v), y;
  y = __builtin_amdgcn_update_dpp(0, x, 0xB1,  0xF, 0xF, true);
  x = __float_as_int(fmaxf(__int_as_float(x), __int_as_float(y)));
  y = __builtin_amdgcn_update_dpp(0, x, 0x4E,  0xF, 0xF, true);
  x = __float_as_int(fmaxf(__int_as_float(x), __int_as_float(y)));
  y = __builtin_amdgcn_update_dpp(0, x, 0x141, 0xF, 0xF, true);
  x = __float_as_int(fmaxf(__int_as_float(x), __int_as_float(y)));
  y = __builtin_amdgcn_update_dpp(0, x, 0x140, 0xF, 0xF, true);
  x = __float_as_int(fmaxf(__int_as_float(x), __int_as_float(y)));
  return __int_as_float(x);
}

__global__ __launch_bounds__(128)
void fa_swin_sink_kernel(const float* __restrict__ Q, const float* __restrict__ K,
                         const float* __restrict__ V, const float* __restrict__ S,
                         float* __restrict__ O) {
  // K tile row-major [key][d], stride 66 (conflict-free B gathers)
  __shared__ unsigned short Ks[32][66];
  // V tile transposed [d][key], stride 34 (contiguous PV B-fragment pairs)
  __shared__ unsigned short Vt[64][34];
  // P (softmax weights) per wave, [row][key], stride 34
  __shared__ unsigned short Ps[4][16][34];

  const int q0   = blockIdx.x * 16;      // query tile base
  const int h    = blockIdx.y;           // kv head
  const int tid  = threadIdx.x;
  const int mq   = tid >> 5;             // wave id == q_mult index
  const int lane = tid & 31;
  const int half = lane >> 4;
  const int col  = lane & 15;

  // ---- Q fragment (A layout), pre-scaled into the exp2 domain ----
  v8u qa_u[2];
  const float* qrow = Q + (((size_t)(q0 + col) * NH + h) * QM + mq) * DH;
  #pragma unroll
  for (int s = 0; s < 2; ++s) {
    #pragma unroll
    for (int i = 0; i < 8; ++i) {
      int d = 32 * s + a_kidx(i, half);
      float2 q2 = *(const float2*)(qrow + d);
      qa_u[s][i] = pk2bf(q2.x * (SM_SCALE * LOG2E), q2.y * (SM_SCALE * LOG2E));
    }
  }
  const v16bf qa0 = __builtin_bit_cast(v16bf, qa_u[0]);
  const v16bf qa1 = __builtin_bit_cast(v16bf, qa_u[1]);

  // all-ones bf16 B matrix for WMMA-based row sums
  v8u ones_u;
  #pragma unroll
  for (int i = 0; i < 8; ++i) ones_u[i] = 0x3f803f80u;
  const v16bf ones = __builtin_bit_cast(v16bf, ones_u);

  // ---- online softmax state (rows r+8*half replicated across 16 lanes/half) ----
  float mrow[8], lrow[8];
  v8f acc[4] = {};
  #pragma unroll
  for (int r = 0; r < 8; ++r) { mrow[r] = -1e30f; lrow[r] = 0.0f; }

  // key-tile range covering the union of sliding windows of rows q0..q0+15
  const int t0 = (q0 >= SW) ? ((q0 - SW + 1) >> 5) : 0;
  const int t1 = (q0 + 15) >> 5;

  for (int kt = t0; kt <= t1; ++kt) {
    const int kb = kt << 5;
    __syncthreads();

    // ---- cooperative stage: 128 threads load 32x64 K and V, convert to bf16 ----
    {
      const int srow = tid >> 2;             // token within tile
      const int sd   = (tid & 3) << 4;       // starting d (16 floats per thread)
      const float* kp = K + ((size_t)(kb + srow) * NH + h) * DH + sd;
      const float* vp = V + ((size_t)(kb + srow) * NH + h) * DH + sd;
      #pragma unroll
      for (int j = 0; j < 4; ++j) {
        float4 kv = *(const float4*)(kp + 4 * j);
        *(unsigned*)&Ks[srow][sd + 4 * j + 0] = pk2bf(kv.x, kv.y);
        *(unsigned*)&Ks[srow][sd + 4 * j + 2] = pk2bf(kv.z, kv.w);
        float4 vv = *(const float4*)(vp + 4 * j);
        Vt[sd + 4 * j + 0][srow] = f2bf(vv.x);
        Vt[sd + 4 * j + 1][srow] = f2bf(vv.y);
        Vt[sd + 4 * j + 2][srow] = f2bf(vv.z);
        Vt[sd + 4 * j + 3][srow] = f2bf(vv.w);
      }
      if (kt < t1) {  // gfx1250 global_prefetch_b8 for the next tile
        __builtin_prefetch(kp + (size_t)32 * NH * DH, 0, 0);
        __builtin_prefetch(vp + (size_t)32 * NH * DH, 0, 0);
      }
    }
    __syncthreads();

    // ---- QK^T: two 16x16 score subtiles (keys kb+16n .. kb+16n+15) ----
    float sc[2][8];
    #pragma unroll
    for (int n = 0; n < 2; ++n) {
      v8u b0, b1;
      #pragma unroll
      for (int i = 0; i < 8; ++i) {
        const int key = 16 * n + col;          // B lane = N = key
        const int d   = a_kidx(i, half);       // B K-dim = d within slice
        b0[i] = ld2(&Ks[key][d]);
        b1[i] = ld2(&Ks[key][32 + d]);
      }
      v8f s8 = {};
      s8 = __builtin_amdgcn_wmma_f32_16x16x32_bf16(
          false, qa0, false, __builtin_bit_cast(v16bf, b0), (short)0, s8, false, false);
      s8 = __builtin_amdgcn_wmma_f32_16x16x32_bf16(
          false, qa1, false, __builtin_bit_cast(v16bf, b1), (short)0, s8, false, false);
      // sliding-window mask as a single unsigned range check per element
      const int d0 = q0 + 8 * half - (kb + 16 * n + col);   // (q - k) - r
      #pragma unroll
      for (int r = 0; r < 8; ++r) {
        sc[n][r] = ((unsigned)(d0 + r) < (unsigned)SW) ? s8[r] : -1e30f;
      }
    }

    // ---- online softmax: row max (DPP all-reduce), exponentials ----
    float scale[8];
    #pragma unroll
    for (int r = 0; r < 8; ++r) {
      const float v  = dpp_max16(fmaxf(sc[0][r], sc[1][r]));
      const float mn = fmaxf(mrow[r], v);
      scale[r] = __builtin_amdgcn_exp2f(mrow[r] - mn);
      mrow[r]  = mn;
      sc[0][r] = __builtin_amdgcn_exp2f(sc[0][r] - mn);
      sc[1][r] = __builtin_amdgcn_exp2f(sc[1][r] - mn);
    }
    #pragma unroll
    for (int t = 0; t < 4; ++t)
      #pragma unroll
      for (int r = 0; r < 8; ++r)
        acc[t][r] *= scale[r];

    // ---- P: C layout -> LDS -> A layout (same-wave round trip) ----
    #pragma unroll
    for (int n = 0; n < 2; ++n)
      #pragma unroll
      for (int r = 0; r < 8; ++r)
        Ps[mq][r + 8 * half][16 * n + col] = f2bf(sc[n][r]);

    asm volatile("s_wait_dscnt 0" ::: "memory");

    v8u pa_u;
    #pragma unroll
    for (int i = 0; i < 8; ++i)
      pa_u[i] = ld2(&Ps[mq][col][a_kidx(i, half)]);
    const v16bf pa = __builtin_bit_cast(v16bf, pa_u);

    // ---- row sums via WMMA against ones: every column of D = row sum ----
    {
      v8f rs = {};
      rs = __builtin_amdgcn_wmma_f32_16x16x32_bf16(
          false, pa, false, ones, (short)0, rs, false, false);
      #pragma unroll
      for (int r = 0; r < 8; ++r)
        lrow[r] = lrow[r] * scale[r] + rs[r];
    }

    // ---- PV: four 16-wide d column tiles, contraction over 32 keys ----
    #pragma unroll
    for (int t = 0; t < 4; ++t) {
      v8u bv;
      #pragma unroll
      for (int i = 0; i < 8; ++i)
        bv[i] = ld2(&Vt[16 * t + col][a_kidx(i, half)]);
      acc[t] = __builtin_amdgcn_wmma_f32_16x16x32_bf16(
          false, pa, false, __builtin_bit_cast(v16bf, bv), (short)0, acc[t], false, false);
    }
  }

  // ---- sink logit joins the denominator only; then normalize and store ----
  const float ssink2 = S[h * QM + mq] * LOG2E;   // same exp2 domain as scores
  float invn[8];
  #pragma unroll
  for (int r = 0; r < 8; ++r) {
    const float mn  = fmaxf(mrow[r], ssink2);
    const float sc2 = __builtin_amdgcn_exp2f(mrow[r] - mn);
    const float den = lrow[r] * sc2 + __builtin_amdgcn_exp2f(ssink2 - mn);
    invn[r] = sc2 * __builtin_amdgcn_rcpf(den);  // folds acc rescale + 1/denominator
  }
  #pragma unroll
  for (int t = 0; t < 4; ++t)
    #pragma unroll
    for (int r = 0; r < 8; ++r) {
      const size_t off = (size_t)(q0 + r + 8 * half) * (NH * QM * DH)
                       + (size_t)(h * QM + mq) * DH + 16 * t + col;
      O[off] = acc[t][r] * invn[r];
    }
}

extern "C" void kernel_launch(void* const* d_in, const int* in_sizes, int n_in,
                              void* d_out, int out_size, void* d_ws, size_t ws_size,
                              hipStream_t stream) {
  const float* Q = (const float*)d_in[0];
  const float* K = (const float*)d_in[1];
  const float* V = (const float*)d_in[2];
  const float* S = (const float*)d_in[3];
  float* O = (float*)d_out;
  dim3 grid(NT / 16, NH);
  fa_swin_sink_kernel<<<grid, 128, 0, stream>>>(Q, K, V, S, O);
}
